// LINEAR_TO_COS_SIM_49117245997167
// MI455X (gfx1250) — matrix operand order repaired and checked by reading, not compiled, not run on hardware
//
#include <hip/hip_runtime.h>
#include <stdint.h>

#define B_ROWS 16384
#define N_COLS 4096
#define K_DIM  512

#define BLK_M 128
#define BLK_N 64
#define TILE_K 32
#define NK (K_DIM / TILE_K)  // 16

#define LDS_STRIDE_F 36   // fused path: fp32 tiles, 144B pitch
#define LDS_PITCH_BF 40   // fast path: bf16 tiles, 80B pitch (16B aligned)

typedef __attribute__((ext_vector_type(16))) __bf16 v16bf;
typedef __attribute__((ext_vector_type(8)))  float  v8f;
typedef int v4i __attribute__((vector_size(16)));
typedef v4i __attribute__((address_space(1))) v4i_g;   // global
typedef v4i __attribute__((address_space(3))) v4i_l;   // LDS

#if defined(__HIP_DEVICE_COMPILE__) && __has_builtin(__builtin_amdgcn_global_load_async_to_lds_b128)
#define USE_ASYNC_LDS 1
#else
#define USE_ASYNC_LDS 0
#endif

__device__ __forceinline__ void wait_async_all() {
#if USE_ASYNC_LDS
#if __has_builtin(__builtin_amdgcn_s_wait_asynccnt)
  __builtin_amdgcn_s_wait_asynccnt(0);
#else
  asm volatile("s_wait_asynccnt 0" ::: "memory");
#endif
#endif
}

__device__ __forceinline__ void copy16B_g2l(const void* gp, void* lp) {
#if USE_ASYNC_LDS
  __builtin_amdgcn_global_load_async_to_lds_b128((v4i_g*)gp, (v4i_l*)lp, 0, 0);
#else
  *(float4*)lp = *(const float4*)gp;
#endif
}

// ============ Kernel 1a: norms only (fallback path) ==========================
__global__ __launch_bounds__(256) void LIN_COS_norms_kernel(
    const float* __restrict__ x, const float* __restrict__ w,
    float* __restrict__ invx, float* __restrict__ invw) {
  int gwave = (blockIdx.x * 256 + threadIdx.x) >> 5;
  int lane  = threadIdx.x & 31;
  const float* src; float* dst; int row;
  if (gwave < B_ROWS) { src = x; dst = invx; row = gwave; }
  else                { src = w; dst = invw; row = gwave - B_ROWS; }
  const float* p = src + (size_t)row * K_DIM;
  float s = 0.f;
#pragma unroll
  for (int i = 0; i < K_DIM / 32; ++i) { float v = p[lane + i * 32]; s += v * v; }
#pragma unroll
  for (int off = 16; off > 0; off >>= 1) s += __shfl_xor(s, off, 32);
  if (lane == 0) dst[row] = 1.0f / fmaxf(sqrtf(s), 1e-8f);
}

// ============ Kernel 1b: norms + bf16 hi/lo packing (fast path) ==============
// One wave per row; each lane handles 8 pairs of consecutive elements.
__global__ __launch_bounds__(256) void LIN_COS_pack_kernel(
    const float* __restrict__ x, const float* __restrict__ w,
    float* __restrict__ invx, float* __restrict__ invw,
    unsigned int* __restrict__ xh, unsigned int* __restrict__ xl,
    unsigned int* __restrict__ wh, unsigned int* __restrict__ wl) {
  int gwave = (blockIdx.x * 256 + threadIdx.x) >> 5;
  int lane  = threadIdx.x & 31;
  const float* src; float* dst; unsigned int *ho, *lo_; int row;
  if (gwave < B_ROWS) { src = x; dst = invx; ho = xh; lo_ = xl; row = gwave; }
  else { src = w; dst = invw; ho = wh; lo_ = wl; row = gwave - B_ROWS; }
  const float2* p2 = (const float2*)(src + (size_t)row * K_DIM);
  unsigned int* hrow = ho  + (size_t)row * (K_DIM / 2);
  unsigned int* lrow = lo_ + (size_t)row * (K_DIM / 2);
  float s = 0.f;
#pragma unroll
  for (int i = 0; i < 8; ++i) {
    int p = lane + 32 * i;           // pair index 0..255
    float2 f = p2[p];
    __bf16 h0 = (__bf16)f.x, h1 = (__bf16)f.y;
    __bf16 l0 = (__bf16)(f.x - (float)h0), l1 = (__bf16)(f.y - (float)h1);
    union { __bf16 b[2]; unsigned int u; } ph, pl;
    ph.b[0] = h0; ph.b[1] = h1; pl.b[0] = l0; pl.b[1] = l1;
    hrow[p] = ph.u; lrow[p] = pl.u;
    s += f.x * f.x + f.y * f.y;
  }
#pragma unroll
  for (int off = 16; off > 0; off >>= 1) s += __shfl_xor(s, off, 32);
  if (lane == 0) dst[row] = 1.0f / fmaxf(sqrtf(s), 1e-8f);
}

// ============ Fast GEMM: pre-split bf16 tiles, conversion-free inner loop ====
template <int ROWS>  // copy ROWS x 32 bf16 (row stride K_DIM) into LDS pitch 40
__device__ __forceinline__ void copy_bf_tile(__bf16* __restrict__ dst,
                                             const __bf16* __restrict__ srcg,
                                             int kk, int tid) {
#pragma unroll
  for (int it = 0; it < ROWS / 64; ++it) {
    int seg = tid + it * 256;        // 16B = 8 bf16 per segment; 4 per row
    int r = seg >> 2;
    int c = (seg & 3) * 8;
    copy16B_g2l(srcg + (size_t)r * K_DIM + kk + c, dst + r * LDS_PITCH_BF + c);
  }
}

__device__ __forceinline__ v16bf load_frag(const __bf16* p0, const __bf16* p1) {
  v16bf v;
  *(float4*)&v       = *(const float4*)p0;   // elems 0..7
  *((float4*)&v + 1) = *(const float4*)p1;   // elems 8..15
  return v;
}

__global__ __launch_bounds__(256) void LIN_COS_wmma_pre_kernel(
    const unsigned short* __restrict__ xh_, const unsigned short* __restrict__ xl_,
    const unsigned short* __restrict__ wh_, const unsigned short* __restrict__ wl_,
    const float* __restrict__ invx, const float* __restrict__ invw,
    float* __restrict__ out) {
  __shared__ __align__(16) __bf16 xh[2][BLK_M * LDS_PITCH_BF];  // 20480 B
  __shared__ __align__(16) __bf16 xl[2][BLK_M * LDS_PITCH_BF];  // 20480 B
  __shared__ __align__(16) __bf16 wh[2][BLK_N * LDS_PITCH_BF];  // 10240 B
  __shared__ __align__(16) __bf16 wl[2][BLK_N * LDS_PITCH_BF];  // 10240 B
  __shared__ float sx[BLK_M];
  __shared__ float sw[BLK_N];

  const int tid   = threadIdx.x;
  const int wave  = tid >> 5;
  const int lane  = tid & 31;
  const int waveM = wave >> 1;   // 0..3
  const int waveN = wave & 1;    // 0..1
  const int half  = lane >> 4;
  const int l     = lane & 15;

  const int rowBase = blockIdx.y * BLK_M;
  const int colBase = blockIdx.x * BLK_N;
  const __bf16* xhg = (const __bf16*)xh_ + (size_t)rowBase * K_DIM;
  const __bf16* xlg = (const __bf16*)xl_ + (size_t)rowBase * K_DIM;
  const __bf16* whg = (const __bf16*)wh_ + (size_t)colBase * K_DIM;
  const __bf16* wlg = (const __bf16*)wl_ + (size_t)colBase * K_DIM;

  if (tid < BLK_M) sx[tid] = invx[rowBase + tid];
  if (tid < BLK_N) sw[tid] = invw[colBase + tid];

  v8f acc[2][2] = {};

  copy_bf_tile<BLK_M>(xh[0], xhg, 0, tid);
  copy_bf_tile<BLK_M>(xl[0], xlg, 0, tid);
  copy_bf_tile<BLK_N>(wh[0], whg, 0, tid);
  copy_bf_tile<BLK_N>(wl[0], wlg, 0, tid);

  for (int ks = 0; ks < NK; ++ks) {
    wait_async_all();
    __syncthreads();
    const int cur = ks & 1;
    if (ks + 1 < NK) {
      const int kk = (ks + 1) * TILE_K;
      copy_bf_tile<BLK_M>(xh[cur ^ 1], xhg, kk, tid);
      copy_bf_tile<BLK_M>(xl[cur ^ 1], xlg, kk, tid);
      copy_bf_tile<BLK_N>(wh[cur ^ 1], whg, kk, tid);
      copy_bf_tile<BLK_N>(wl[cur ^ 1], wlg, kk, tid);
    }

    // A fragments: lanes 0-15: K=0..7,16..23; lanes 16-31: +8
    v16bf ah[2], al[2], bh[2], bl[2];
    const int kA0 = half * 8;
#pragma unroll
    for (int mt = 0; mt < 2; ++mt) {
      const int base = (waveM * 32 + mt * 16 + l) * LDS_PITCH_BF + kA0;
      ah[mt] = load_frag(&xh[cur][base], &xh[cur][base + 16]);
      al[mt] = load_frag(&xl[cur][base], &xl[cur][base + 16]);
    }
    // B fragments: lanes 0-15: K=0..7,16..23; lanes 16-31: +8
    const int kB0 = half * 8;
#pragma unroll
    for (int nt = 0; nt < 2; ++nt) {
      const int base = (waveN * 32 + nt * 16 + l) * LDS_PITCH_BF + kB0;
      bh[nt] = load_frag(&wh[cur][base], &wh[cur][base + 16]);
      bl[nt] = load_frag(&wl[cur][base], &wl[cur][base + 16]);
    }

#pragma unroll
    for (int mt = 0; mt < 2; ++mt)
#pragma unroll
      for (int nt = 0; nt < 2; ++nt) {
        acc[mt][nt] = __builtin_amdgcn_wmma_f32_16x16x32_bf16(
            false, ah[mt], false, bh[nt], (short)0, acc[mt][nt], false, false);
        acc[mt][nt] = __builtin_amdgcn_wmma_f32_16x16x32_bf16(
            false, al[mt], false, bh[nt], (short)0, acc[mt][nt], false, false);
        acc[mt][nt] = __builtin_amdgcn_wmma_f32_16x16x32_bf16(
            false, ah[mt], false, bl[nt], (short)0, acc[mt][nt], false, false);
      }
  }

#pragma unroll
  for (int mt = 0; mt < 2; ++mt) {
    const int m0 = waveM * 32 + mt * 16 + half * 8;
#pragma unroll
    for (int nt = 0; nt < 2; ++nt) {
      const int cl = waveN * 32 + nt * 16 + l;
      const float swv = sw[cl];
      const size_t col = (size_t)colBase + cl;
#pragma unroll
      for (int j = 0; j < 8; ++j) {
        const int rl = m0 + j;
        out[(size_t)(rowBase + rl) * N_COLS + col] = acc[mt][nt][j] * sx[rl] * swv;
      }
    }
  }
}

// ============ Fallback GEMM: fused f32->bf16 split in inner loop =============
template <int ROWS>
__device__ __forceinline__ void copy_f32_tile(float* __restrict__ dst,
                                              const float* __restrict__ srcg,
                                              int kk, int tid) {
#pragma unroll
  for (int it = 0; it < ROWS / 32; ++it) {
    int seg = tid + it * 256;
    int r = seg >> 3;
    int c = (seg & 7) * 4;
    copy16B_g2l(srcg + (size_t)r * K_DIM + kk + c, dst + r * LDS_STRIDE_F + c);
  }
}

__device__ __forceinline__ void split_bf16x16(const float4& q0, const float4& q1,
                                              const float4& q2, const float4& q3,
                                              v16bf& hi, v16bf& lo) {
  float f[16] = {q0.x, q0.y, q0.z, q0.w, q1.x, q1.y, q1.z, q1.w,
                 q2.x, q2.y, q2.z, q2.w, q3.x, q3.y, q3.z, q3.w};
#pragma unroll
  for (int i = 0; i < 16; ++i) {
    __bf16 h = (__bf16)f[i];
    hi[i] = h;
    lo[i] = (__bf16)(f[i] - (float)h);
  }
}

__global__ __launch_bounds__(256) void LIN_COS_wmma_fused_kernel(
    const float* __restrict__ x, const float* __restrict__ w,
    const float* __restrict__ invx, const float* __restrict__ invw,
    float* __restrict__ out) {
  __shared__ __align__(16) float xs[2][BLK_M * LDS_STRIDE_F];
  __shared__ __align__(16) float wt[2][BLK_N * LDS_STRIDE_F];
  __shared__ float sx[BLK_M];
  __shared__ float sw[BLK_N];

  const int tid   = threadIdx.x;
  const int wave  = tid >> 5;
  const int lane  = tid & 31;
  const int waveM = wave >> 1;
  const int waveN = wave & 1;
  const int half  = lane >> 4;
  const int l     = lane & 15;

  const int rowBase = blockIdx.y * BLK_M;
  const int colBase = blockIdx.x * BLK_N;
  const float* xblk = x + (size_t)rowBase * K_DIM;
  const float* wblk = w + (size_t)colBase * K_DIM;

  if (tid < BLK_M) sx[tid] = invx[rowBase + tid];
  if (tid < BLK_N) sw[tid] = invw[colBase + tid];

  v8f acc[2][2] = {};
  copy_f32_tile<BLK_M>(xs[0], xblk, 0, tid);
  copy_f32_tile<BLK_N>(wt[0], wblk, 0, tid);

  for (int ks = 0; ks < NK; ++ks) {
    wait_async_all();
    __syncthreads();
    const int cur = ks & 1;
    if (ks + 1 < NK) {
      copy_f32_tile<BLK_M>(xs[cur ^ 1], xblk, (ks + 1) * TILE_K, tid);
      copy_f32_tile<BLK_N>(wt[cur ^ 1], wblk, (ks + 1) * TILE_K, tid);
    }
    const float* X = xs[cur];
    const float* W = wt[cur];

    v16bf ah[2], al[2], bh[2], bl[2];
    const int kA0 = half * 8;
#pragma unroll
    for (int mt = 0; mt < 2; ++mt) {
      const float* rp = X + (waveM * 32 + mt * 16 + l) * LDS_STRIDE_F + kA0;
      split_bf16x16(*(const float4*)(rp), *(const float4*)(rp + 4),
                    *(const float4*)(rp + 16), *(const float4*)(rp + 20),
                    ah[mt], al[mt]);
    }
    const int kB0 = half * 8;
#pragma unroll
    for (int nt = 0; nt < 2; ++nt) {
      const float* rp = W + (waveN * 32 + nt * 16 + l) * LDS_STRIDE_F + kB0;
      split_bf16x16(*(const float4*)(rp), *(const float4*)(rp + 4),
                    *(const float4*)(rp + 16), *(const float4*)(rp + 20),
                    bh[nt], bl[nt]);
    }

#pragma unroll
    for (int mt = 0; mt < 2; ++mt)
#pragma unroll
      for (int nt = 0; nt < 2; ++nt) {
        acc[mt][nt] = __builtin_amdgcn_wmma_f32_16x16x32_bf16(
            false, ah[mt], false, bh[nt], (short)0, acc[mt][nt], false, false);
        acc[mt][nt] = __builtin_amdgcn_wmma_f32_16x16x32_bf16(
            false, al[mt], false, bh[nt], (short)0, acc[mt][nt], false, false);
        acc[mt][nt] = __builtin_amdgcn_wmma_f32_16x16x32_bf16(
            false, ah[mt], false, bl[nt], (short)0, acc[mt][nt], false, false);
      }
  }

#pragma unroll
  for (int mt = 0; mt < 2; ++mt) {
    const int m0 = waveM * 32 + mt * 16 + half * 8;
#pragma unroll
    for (int nt = 0; nt < 2; ++nt) {
      const int cl = waveN * 32 + nt * 16 + l;
      const float swv = sw[cl];
      const size_t col = (size_t)colBase + cl;
#pragma unroll
      for (int j = 0; j < 8; ++j) {
        const int rl = m0 + j;
        out[(size_t)(rowBase + rl) * N_COLS + col] = acc[mt][nt][j] * sx[rl] * swv;
      }
    }
  }
}

extern "C" void kernel_launch(void* const* d_in, const int* in_sizes, int n_in,
                              void* d_out, int out_size, void* d_ws, size_t ws_size,
                              hipStream_t stream) {
  const float* x = (const float*)d_in[0];
  const float* w = (const float*)d_in[1];
  float* out = (float*)d_out;

  char* ws = (char*)d_ws;
  float* invx = (float*)ws;                               // 64 KB
  float* invw = (float*)(ws + 65536);                     // 16 KB
  const size_t xhOff = 81920;
  const size_t xBytes = (size_t)B_ROWS * K_DIM * 2;       // 16 MB
  const size_t wBytes = (size_t)N_COLS * K_DIM * 2;       // 4 MB
  const size_t need = xhOff + 2 * xBytes + 2 * wBytes;    // ~40.1 MB

  const int normBlocks = (B_ROWS + N_COLS) / 8;  // one wave per row
  dim3 grid(N_COLS / BLK_N, B_ROWS / BLK_M);     // (64, 128)

  if (ws_size >= need) {
    unsigned int* xh = (unsigned int*)(ws + xhOff);
    unsigned int* xl = (unsigned int*)(ws + xhOff + xBytes);
    unsigned int* wh = (unsigned int*)(ws + xhOff + 2 * xBytes);
    unsigned int* wl = (unsigned int*)(ws + xhOff + 2 * xBytes + wBytes);
    LIN_COS_pack_kernel<<<normBlocks, 256, 0, stream>>>(x, w, invx, invw,
                                                        xh, xl, wh, wl);
    LIN_COS_wmma_pre_kernel<<<grid, 256, 0, stream>>>(
        (const unsigned short*)xh, (const unsigned short*)xl,
        (const unsigned short*)wh, (const unsigned short*)wl, invx, invw, out);
  } else {
    LIN_COS_norms_kernel<<<normBlocks, 256, 0, stream>>>(x, w, invx, invw);
    LIN_COS_wmma_fused_kernel<<<grid, 256, 0, stream>>>(x, w, invx, invw, out);
  }
}
